// PTv3ResConvEncoder_5282809774893
// MI455X (gfx1250) — matrix operand order repaired and checked by reading, not compiled, not run on hardware
//
#include <hip/hip_runtime.h>

typedef __attribute__((ext_vector_type(16))) _Float16 v16h;
typedef __attribute__((ext_vector_type(8)))  _Float16 v8h;
typedef __attribute__((ext_vector_type(8)))  float    v8f;

#define GRID_DIM 256
#define PADP     262   // 256 + 2*3 halo
#define BATCH    4
#define NPTS     50000
#define CIN      64
#define COUT     128
#define NT       8     // 128 output channels / 16

// ---- workspace layout (bytes, all 256-aligned) ----
#define OFF_BEV32 0UL                 // [B,256,256,64] fp32 scatter grid      (67,108,864 B)
#define OFF_BEVP  67108864UL          // [B,262,262,64] f16 padded             (35,145,728 B)
#define OFF_H1P   102254592UL         // [B,262,262,128] f16 padded            (70,291,456 B)
#define OFF_BP1   172546048UL         // conv1 packed weights                  (   802,816 B)
#define OFF_BP2   173348864UL         // conv2 packed weights                  ( 1,605,632 B)
#define OFF_BP3   174954496UL         // conv3 packed weights                  (   802,816 B)

static __device__ __forceinline__ float gelu_exact(float x) {
    return 0.5f * x * (1.0f + erff(x * 0.70710678118654752f));
}

// async copy of one 16B chunk: global memory -> LDS, tracked on ASYNCcnt
static __device__ __forceinline__ void async_copy16(const _Float16* gp, _Float16* lp) {
    unsigned lds_off = (unsigned)(unsigned long)lp;   // low 32 bits = LDS offset
    asm volatile("global_load_async_to_lds_b128 %0, %1, off"
                 :: "v"(lds_off), "v"(gp) : "memory");
}
static __device__ __forceinline__ void async_wait0() {
    asm volatile("s_wait_asynccnt 0x0" ::: "memory");
}

// ---------------- zero workspace regions ----------------
__global__ void zero_ws_kernel(uint4* __restrict__ p, long n16) {
    long i = (long)blockIdx.x * blockDim.x + threadIdx.x;
    long stride = (long)gridDim.x * blockDim.x;
    uint4 z = {0u, 0u, 0u, 0u};
    for (; i < n16; i += stride) p[i] = z;
}

// ---------------- scatter points -> BEV fp32 grid [B,H,W,C] ----------------
__global__ void scatter_kernel(const float* __restrict__ coords,
                               const float* __restrict__ feats,
                               float* __restrict__ grid) {
    long tid = (long)blockIdx.x * blockDim.x + threadIdx.x;
    const long total = (long)BATCH * NPTS * 8;
    if (tid >= total) return;
    int  cc = (int)(tid & 7);        // 8-channel chunk
    long pi = tid >> 3;              // point index in [0, B*N)
    int  b  = (int)(pi / NPTS);
    long c3 = pi * 3;
    float px = coords[c3 + 0];
    float py = coords[c3 + 1];
    float pz = coords[c3 + 2];
    int xi = (int)floorf((px + 51.2f) * 2.5f);
    int yi = (int)floorf((py + 51.2f) * 2.5f);
    int zi = (int)floorf((pz + 10.0f) * 0.05f);
    if (xi < 0 || xi >= GRID_DIM || yi < 0 || yi >= GRID_DIM || zi != 0) return;
    long gbase = (((long)b * GRID_DIM + yi) * GRID_DIM + xi) * CIN + cc * 8;
    long fbase = pi * CIN + cc * 8;
    #pragma unroll
    for (int c = 0; c < 8; ++c)
        atomicAdd(&grid[gbase + c], feats[fbase + c]);
}

// ---------------- fp32 grid -> padded NHWC f16 ----------------
__global__ void convert_kernel(const float* __restrict__ grid,
                               _Float16* __restrict__ bevp) {
    long i = (long)blockIdx.x * blockDim.x + threadIdx.x;
    long stride = (long)gridDim.x * blockDim.x;
    const long total = (long)BATCH * GRID_DIM * GRID_DIM * CIN;
    for (; i < total; i += stride) {
        int c = (int)(i & (CIN - 1));
        int x = (int)((i >> 6) & 255);
        int y = (int)((i >> 14) & 255);
        int b = (int)(i >> 22);
        bevp[(((long)(b * PADP + y + 3)) * PADP + (x + 3)) * CIN + c] = (_Float16)grid[i];
    }
}

// ---------------- pack conv weights [O,IC,7,7] fp32 -> WMMA B-fragment order ----------------
// layout: [tap][kb][nt][lane][16 x f16]; lane<16 -> K {0..7,16..23}, lane>=16 -> K {8..15,24..31} (of the k-block)
__global__ void prepack_kernel(const float* __restrict__ w,
                               _Float16* __restrict__ bp, int ic, int kb_count) {
    long idx = (long)blockIdx.x * blockDim.x + threadIdx.x;
    long total = 49L * kb_count * NT * 512;
    if (idx >= total) return;
    int e    = (int)(idx & 15);
    int lane = (int)((idx >> 4) & 31);
    int nt   = (int)((idx >> 9) & 7);
    long r   = idx >> 12;            // tap*kb_count + kb
    int kb   = (int)(r % kb_count);
    int tap  = (int)(r / kb_count);
    int n = nt * 16 + (lane & 15);
    int k = kb * 32 + ((lane >= 16) ? 8 : 0) + (e & 7) + ((e >= 8) ? 16 : 0);
    int ky = tap / 7, kx = tap % 7;
    bp[idx] = (_Float16)w[(((long)n * ic + k) * 7 + ky) * 7 + kx];
}

// ---------------- implicit-GEMM accumulation ----------------
// A rows double-buffered in LDS; the NEXT row is written by GLOBAL_LOAD_ASYNC_TO_LDS_B128
// (ASYNCcnt) while the current row is consumed, so row staging never touches VGPRs or
// LOADcnt. Inner (kx,kb) loop fully unrolled with one-step register prefetch of A (ds)
// and the four B fragments (global/L2), so the WMMA pipe never waits on a just-issued load.
template <int IC, int KB>
static __device__ __forceinline__ void accum_branch_lds(
    const _Float16* __restrict__ Ain, const _Float16* __restrict__ Bp,
    _Float16* smem, int b, int y, int xbb, int wm, int wn, int lane, int tid,
    v8f acc[4]) {
    constexpr int CS    = IC + 8;              // LDS column stride (halfs)
    constexpr int CH8   = IC / 8;              // 16B chunks per column
    constexpr int NCH   = 70 * CH8;            // chunks per row patch
    constexpr int NREG  = (NCH + 255) / 256;   // chunks per thread
    constexpr int BUFSZ = 70 * CS;             // halfs per row buffer
    constexpr int T     = 7 * KB;              // pipelined steps per row
    const int klo = (lane >= 16) ? 8 : 0;      // A frag: lane<16 K{0..7,16..23}, lane>=16 K{8..15,24..31}
    const int xl  = lane & 15;
    const int abase0 = (wm * 16 + xl) * CS + klo;
    const long rowstride = (long)PADP * IC;
    const _Float16* src0 = Ain + ((long)(b * PADP + y) * PADP + xbb) * IC;

    // ---- prologue: async-stage row 0 into buffer 0 ----
    __syncthreads();                           // prior users of smem are done
    #pragma unroll
    for (int r = 0; r < NREG; ++r) {
        int t = tid + r * 256;
        if (t < NCH) {
            int col = t / CH8, g = t - col * CH8;
            async_copy16(src0 + (long)col * IC + g * 8, &smem[col * CS + g * 8]);
        }
    }
    async_wait0();
    __syncthreads();

    int cur = 0;
    #pragma unroll 1
    for (int ky = 0; ky < 7; ++ky) {
        // async-prefetch the next row directly into the other LDS buffer;
        // transfer runs on ASYNCcnt while the WMMAs below execute
        if (ky < 6) {
            const _Float16* src = src0 + (long)(ky + 1) * rowstride;
            _Float16* nbuf = smem + (cur ^ 1) * BUFSZ;
            #pragma unroll
            for (int r = 0; r < NREG; ++r) {
                int t = tid + r * 256;
                if (t < NCH) {
                    int col = t / CH8, g = t - col * CH8;
                    async_copy16(src + (long)col * IC + g * 8, &nbuf[col * CS + g * 8]);
                }
            }
        }
        // ---- compute on buffer `cur`, software-pipelined over t = (kx,kb) ----
        const _Float16* buf = smem + cur * BUFSZ;
        const _Float16* bky = Bp + (long)(ky * T) * (NT * 512) +
                              (long)(wn * 4) * 512 + (long)lane * 16;
        union AU { v16h v; v8h h[2]; };
        AU a_cur, a_nxt;
        v16h bc0, bc1, bc2, bc3;
        {
            const _Float16* ap = &buf[abase0];
            a_cur.h[0] = *(const v8h*)ap;
            a_cur.h[1] = *(const v8h*)(ap + 16);
            bc0 = *(const v16h*)(bky);
            bc1 = *(const v16h*)(bky + 512);
            bc2 = *(const v16h*)(bky + 1024);
            bc3 = *(const v16h*)(bky + 1536);
        }
        #pragma unroll
        for (int t = 0; t < T; ++t) {
            v16h bn0, bn1, bn2, bn3;
            if (t + 1 < T) {
                int kx = (t + 1) / KB, kb = (t + 1) - kx * KB;
                const _Float16* ap = &buf[abase0 + kx * CS + kb * 32];
                a_nxt.h[0] = *(const v8h*)ap;
                a_nxt.h[1] = *(const v8h*)(ap + 16);
                const _Float16* p = bky + (long)(t + 1) * (NT * 512);
                bn0 = *(const v16h*)(p);
                bn1 = *(const v16h*)(p + 512);
                bn2 = *(const v16h*)(p + 1024);
                bn3 = *(const v16h*)(p + 1536);
            }
            acc[0] = __builtin_amdgcn_wmma_f32_16x16x32_f16(
                false, a_cur.v, false, bc0, (short)0, acc[0], false, false);
            acc[1] = __builtin_amdgcn_wmma_f32_16x16x32_f16(
                false, a_cur.v, false, bc1, (short)0, acc[1], false, false);
            acc[2] = __builtin_amdgcn_wmma_f32_16x16x32_f16(
                false, a_cur.v, false, bc2, (short)0, acc[2], false, false);
            acc[3] = __builtin_amdgcn_wmma_f32_16x16x32_f16(
                false, a_cur.v, false, bc3, (short)0, acc[3], false, false);
            if (t + 1 < T) {
                a_cur = a_nxt;
                bc0 = bn0; bc1 = bn1; bc2 = bn2; bc3 = bn3;
            }
        }
        // ---- retire the async row transfer, flip buffers ----
        if (ky < 6) {
            async_wait0();
            cur ^= 1;
        }
        __syncthreads();
    }
}

// ---------------- conv1: h1 = GELU(BN1(conv7x7(bev))) -> padded f16 ----------------
__global__ __launch_bounds__(256)
void conv1_kernel(const _Float16* __restrict__ bevp, const _Float16* __restrict__ bp1,
                  const float* __restrict__ bias, const float* __restrict__ g,
                  const float* __restrict__ be, const float* __restrict__ mean,
                  const float* __restrict__ var, _Float16* __restrict__ h1p) {
    __shared__ __align__(32) _Float16 smem[2 * 70 * (CIN + 8)];
    const int tid  = threadIdx.x;
    const int lane = tid & 31;
    const int w    = tid >> 5;
    const int wm   = w & 3, wn = w >> 2;
    const int b    = blockIdx.y;
    const int m0   = blockIdx.x * 64 + wm * 16;
    const int y    = m0 >> 8, xb = m0 & 255;
    const int xbb  = blockIdx.x * 64 & 255;    // block-level column base
    v8f acc[4] = {};
    accum_branch_lds<CIN, 2>(bevp, bp1, smem, b, y, xbb, wm, wn, lane, tid, acc);
    const int mhi = (lane >= 16) ? 8 : 0;
    #pragma unroll
    for (int j = 0; j < 4; ++j) {
        int o = (wn * 4 + j) * 16 + (lane & 15);
        float inv  = g[o] * rsqrtf(var[o] + 1e-5f);
        float cadd = be[o] + (bias[o] - mean[o]) * inv;
        #pragma unroll
        for (int r = 0; r < 8; ++r) {
            int mm = mhi + r;
            float vout = gelu_exact(acc[j][r] * inv + cadd);
            h1p[(((long)(b * PADP + y + 3)) * PADP + (xb + mm + 3)) * COUT + o] =
                (_Float16)vout;
        }
    }
}

// ---------------- fused: out = GELU(BN2(conv(h1)) + BN3(conv(bev))) -> fp32 NCHW ----------------
__global__ __launch_bounds__(256)
void conv23_kernel(const _Float16* __restrict__ h1p, const _Float16* __restrict__ bp2,
                   const _Float16* __restrict__ bevp, const _Float16* __restrict__ bp3,
                   const float* __restrict__ b2, const float* __restrict__ g2,
                   const float* __restrict__ be2, const float* __restrict__ m2,
                   const float* __restrict__ v2,
                   const float* __restrict__ b3, const float* __restrict__ g3,
                   const float* __restrict__ be3, const float* __restrict__ m3,
                   const float* __restrict__ v3, float* __restrict__ out) {
    __shared__ __align__(32) _Float16 smem[2 * 70 * (COUT + 8)];   // reused by both branches
    const int tid  = threadIdx.x;
    const int lane = tid & 31;
    const int w    = tid >> 5;
    const int wm   = w & 3, wn = w >> 2;
    const int b    = blockIdx.y;
    const int m0   = blockIdx.x * 64 + wm * 16;
    const int y    = m0 >> 8;
    const int xbb  = blockIdx.x * 64 & 255;
    v8f acc2[4] = {};
    v8f acc3[4] = {};
    accum_branch_lds<COUT, 4>(h1p, bp2, smem, b, y, xbb, wm, wn, lane, tid, acc2);
    accum_branch_lds<CIN, 2>(bevp, bp3, smem, b, y, xbb, wm, wn, lane, tid, acc3);
    const int mhi = (lane >= 16) ? 8 : 0;
    #pragma unroll
    for (int j = 0; j < 4; ++j) {
        int o = (wn * 4 + j) * 16 + (lane & 15);
        float inv2 = g2[o] * rsqrtf(v2[o] + 1e-5f);
        float c2   = be2[o] + (b2[o] - m2[o]) * inv2;
        float inv3 = g3[o] * rsqrtf(v3[o] + 1e-5f);
        float c3   = be3[o] + (b3[o] - m3[o]) * inv3;
        #pragma unroll
        for (int r = 0; r < 8; ++r) {
            int mm = mhi + r;
            float pre = acc2[j][r] * inv2 + c2 + acc3[j][r] * inv3 + c3;
            out[((long)(b * COUT + o)) * 65536 + (m0 + mm)] = gelu_exact(pre);
        }
    }
}

extern "C" void kernel_launch(void* const* d_in, const int* in_sizes, int n_in,
                              void* d_out, int out_size, void* d_ws, size_t ws_size,
                              hipStream_t stream) {
    (void)in_sizes; (void)n_in; (void)out_size; (void)ws_size;
    const float* coords = (const float*)d_in[0];
    const float* feats  = (const float*)d_in[1];
    const float* enc_w1 = (const float*)d_in[2];
    const float* enc_b1 = (const float*)d_in[3];
    const float* enc_g1 = (const float*)d_in[4];
    const float* enc_be1= (const float*)d_in[5];
    const float* enc_m1 = (const float*)d_in[6];
    const float* enc_v1 = (const float*)d_in[7];
    const float* enc_w2 = (const float*)d_in[8];
    const float* enc_b2 = (const float*)d_in[9];
    const float* enc_g2 = (const float*)d_in[10];
    const float* enc_be2= (const float*)d_in[11];
    const float* enc_m2 = (const float*)d_in[12];
    const float* enc_v2 = (const float*)d_in[13];
    const float* bot_w  = (const float*)d_in[14];
    const float* bot_b  = (const float*)d_in[15];
    const float* bot_g  = (const float*)d_in[16];
    const float* bot_be = (const float*)d_in[17];
    const float* bot_m  = (const float*)d_in[18];
    const float* bot_v  = (const float*)d_in[19];

    char* ws = (char*)d_ws;
    float*    bev32 = (float*)(ws + OFF_BEV32);
    _Float16* bevp  = (_Float16*)(ws + OFF_BEVP);
    _Float16* h1p   = (_Float16*)(ws + OFF_H1P);
    _Float16* bp1   = (_Float16*)(ws + OFF_BP1);
    _Float16* bp2   = (_Float16*)(ws + OFF_BP2);
    _Float16* bp3   = (_Float16*)(ws + OFF_BP3);

    // zero scatter grid + padded f16 activation buffers (halos must be 0)
    zero_ws_kernel<<<2048, 256, 0, stream>>>((uint4*)ws, (long)(OFF_BP1 / 16));

    // pack weights into WMMA B-fragment order
    prepack_kernel<<<(49 * 2 * NT * 512 + 255) / 256, 256, 0, stream>>>(enc_w1, bp1, CIN, 2);
    prepack_kernel<<<(49 * 4 * NT * 512 + 255) / 256, 256, 0, stream>>>(enc_w2, bp2, COUT, 4);
    prepack_kernel<<<(49 * 2 * NT * 512 + 255) / 256, 256, 0, stream>>>(bot_w, bp3, CIN, 2);

    // scatter-add points into fp32 BEV grid
    scatter_kernel<<<(BATCH * NPTS * 8 + 255) / 256, 256, 0, stream>>>(coords, feats, bev32);

    // convert to padded NHWC f16
    convert_kernel<<<4096, 256, 0, stream>>>(bev32, bevp);

    // conv1 -> BN -> GELU -> padded f16
    dim3 cg(1024, BATCH);
    conv1_kernel<<<cg, 256, 0, stream>>>(bevp, bp1, enc_b1, enc_g1, enc_be1, enc_m1,
                                         enc_v1, h1p);

    // fused conv2 + BN2 + conv3 + BN3 + residual + GELU -> fp32 NCHW output
    conv23_kernel<<<cg, 256, 0, stream>>>(h1p, bp2, bevp, bp3,
                                          enc_b2, enc_g2, enc_be2, enc_m2, enc_v2,
                                          bot_b, bot_g, bot_be, bot_m, bot_v,
                                          (float*)d_out);
}
